// MultiHeadAttention_16793322127561
// MI455X (gfx1250) — matrix operand order repaired and checked by reading, not compiled, not run on hardware
//
#include <hip/hip_runtime.h>

typedef __attribute__((ext_vector_type(16))) _Float16 v16h;
typedef __attribute__((ext_vector_type(8)))  _Float16 v8h;
typedef __attribute__((ext_vector_type(4)))  _Float16 v4h;
typedef __attribute__((ext_vector_type(8)))  float    v8f;
typedef __attribute__((ext_vector_type(4)))  unsigned int u32x4;
typedef __attribute__((ext_vector_type(8)))  int      i32x8;
typedef __attribute__((ext_vector_type(4)))  int      i32x4;

static constexpr int Bb = 4, Qq = 512, Kn = 4096, Dm = 1024, Hh = 8, Dh = 128;

// Build an A/B-style v16h fragment from two contiguous 8-half chunks.
__device__ __forceinline__ v16h make_frag(const _Float16* lo, const _Float16* hi_p) {
    v8h a = *reinterpret_cast<const v8h*>(lo);
    v8h b = *reinterpret_cast<const v8h*>(hi_p);
    v16h r;
#pragma unroll
    for (int e = 0; e < 8; ++e) { r[e] = a[e]; r[8 + e] = b[e]; }
    return r;
}

// A fragment from a padded LDS row: e<8 -> k=e+8*hi, e>=8 -> k=e+8+8*hi
__device__ __forceinline__ v16h make_afrag(const _Float16* row, int hi) {
    return make_frag(row + hi * 8, row + 16 + hi * 8);
}

__device__ __forceinline__ void ld4cvt(const float* p, _Float16* d) {
    float4 v = *reinterpret_cast<const float4*>(p);
    d[0] = (_Float16)v.x; d[1] = (_Float16)v.y; d[2] = (_Float16)v.z; d[3] = (_Float16)v.w;
}
__device__ __forceinline__ void ld4cvt(const _Float16* p, _Float16* d) {
    v4h v = *reinterpret_cast<const v4h*>(p);
    d[0] = v[0]; d[1] = v[1]; d[2] = v[2]; d[3] = v[3];
}

__device__ __forceinline__ void wait_tensorcnt0() {
#if __has_builtin(__builtin_amdgcn_s_wait_tensorcnt)
    __builtin_amdgcn_s_wait_tensorcnt(0);
#else
    asm volatile("s_wait_tensorcnt 0x0" ::: "memory");
#endif
}

// ---------------------------------------------------------------------------
// 16-lane butterfly reductions via DPP16 (no LDS traffic, pure VALU):
//   xor1 = quad_perm(1,0,3,2)=0xB1, xor2 = quad_perm(2,3,0,1)=0x4E,
//   cross-quad = row_half_mirror (0x141), cross-8 = row_mirror (0x140).
// After all four steps every lane of a 16-lane row holds the row reduction.
// CTRL must be an immediate -> template parameter.
// ---------------------------------------------------------------------------
template <int CTRL>
__device__ __forceinline__ float dpp_bfly(float x) {
    int y = __builtin_amdgcn_update_dpp(0, __builtin_bit_cast(int, x), CTRL, 0xF, 0xF, true);
    return __builtin_bit_cast(float, y);
}
__device__ __forceinline__ float red_max16(float x) {
    x = fmaxf(x, dpp_bfly<0xB1>(x));
    x = fmaxf(x, dpp_bfly<0x4E>(x));
    x = fmaxf(x, dpp_bfly<0x141>(x));
    x = fmaxf(x, dpp_bfly<0x140>(x));
    return x;
}
__device__ __forceinline__ float red_add16(float x) {
    x += dpp_bfly<0xB1>(x);
    x += dpp_bfly<0x4E>(x);
    x += dpp_bfly<0x141>(x);
    x += dpp_bfly<0x140>(x);
    return x;
}

// ---------------------------------------------------------------------------
// Generic WMMA GEMM: C[M x N] = A[M x K] * B[K x N] (+ bias).
// Block: 256 threads (8 waves), tile 128x128, K-step 32.
// Wave w -> 32x64 region: A row-strip (w%4)*32, B col-strip (w/4)*64.
// 8 v_wmma per wave per K-step; B fragments reused across both A sub-rows.
// ---------------------------------------------------------------------------
template <typename TA, bool OUT_F16, bool HAS_BIAS>
__global__ void __launch_bounds__(256) gemm_wmma_kernel(
    const TA* __restrict__ A, const float* __restrict__ Bw,
    void* __restrict__ Cout, const float* __restrict__ bias,
    int M, int N, int Kd)
{
    __shared__ _Float16 As [128][40];   // [row][k]  (pad 32->40)
    __shared__ _Float16 BsT[128][40];   // [n][k]    (B stored transposed)

    const int tid  = threadIdx.x;
    const int wave = tid >> 5, lane = tid & 31;
    const int ln   = lane & 15, hi = lane >> 4;
    const int m0 = blockIdx.y * 128, n0 = blockIdx.x * 128;
    const int it = wave & 3;     // 32-row strip
    const int jt = wave >> 2;    // 64-col strip

    v8f acc[2][4];
#pragma unroll
    for (int s = 0; s < 2; ++s)
#pragma unroll
        for (int t = 0; t < 4; ++t) acc[s][t] = (v8f){0,0,0,0,0,0,0,0};

    for (int k0 = 0; k0 < Kd; k0 += 32) {
        // A tile 128x32 -> LDS (f16)
#pragma unroll
        for (int l = 0; l < 4; ++l) {
            int e = (l * 256 + tid) * 4;
            int r = e >> 5, c = e & 31;
            _Float16 t4[4];
            ld4cvt(A + (size_t)(m0 + r) * Kd + k0 + c, t4);
            As[r][c + 0] = t4[0]; As[r][c + 1] = t4[1];
            As[r][c + 2] = t4[2]; As[r][c + 3] = t4[3];
        }
        // B tile 32x128 -> LDS transposed (f16)
#pragma unroll
        for (int l = 0; l < 4; ++l) {
            int e = (l * 256 + tid) * 4;
            int r = e >> 7, c = e & 127;
            float4 v = *reinterpret_cast<const float4*>(Bw + (size_t)(k0 + r) * N + n0 + c);
            BsT[c + 0][r] = (_Float16)v.x; BsT[c + 1][r] = (_Float16)v.y;
            BsT[c + 2][r] = (_Float16)v.z; BsT[c + 3][r] = (_Float16)v.w;
        }
        // Prefetch next K tile into cache while this one is consumed.
        if (k0 + 32 < Kd) {
            __builtin_prefetch(A  + (size_t)(m0 + (tid >> 1)) * Kd + k0 + 32, 0, 1);
            __builtin_prefetch(Bw + (size_t)(k0 + 32 + (tid & 31)) * N + n0 + (tid & 127), 0, 1);
        }
        __syncthreads();

        v16h a0 = make_afrag(&As[it * 32 + ln][0], hi);
        v16h a1 = make_afrag(&As[it * 32 + 16 + ln][0], hi);
#pragma unroll
        for (int t = 0; t < 4; ++t) {
            const _Float16* brow = &BsT[jt * 64 + t * 16 + ln][0];
            v16h bt = make_frag(brow + hi * 8, brow + 16 + hi * 8);
            acc[0][t] = __builtin_amdgcn_wmma_f32_16x16x32_f16(false, a0, false, bt, (short)0, acc[0][t], false, false);
            acc[1][t] = __builtin_amdgcn_wmma_f32_16x16x32_f16(false, a1, false, bt, (short)0, acc[1][t], false, false);
        }
        __syncthreads();
    }

    // D layout: VGPR r -> row r + 8*hi, col = ln
#pragma unroll
    for (int s = 0; s < 2; ++s) {
#pragma unroll
        for (int t = 0; t < 4; ++t) {
#pragma unroll
            for (int r = 0; r < 8; ++r) {
                size_t row = (size_t)(m0 + it * 32 + s * 16 + r + hi * 8);
                int col = n0 + jt * 64 + t * 16 + ln;
                if constexpr (OUT_F16) {
                    ((_Float16*)Cout)[row * N + col] = (_Float16)acc[s][t][r];
                } else {
                    float bv = HAS_BIAS ? bias[col] : 0.f;
                    ((float*)Cout)[row * N + col] = acc[s][t][r] + bv;
                }
            }
        }
    }
}

// ---------------------------------------------------------------------------
// Flash attention: block = 8 waves, each wave owns 16 q rows x DH=128.
// K chunk (32x128 f16) staged via the Tensor Data Mover with LDS padding
// (pad 4 DWORDs every 64 DWORDs -> [32][136] layout); V chunk staged
// transposed by the block so PV B-fragments are two contiguous 16B LDS reads.
// Softmax row reductions are DPP16 butterflies (no ds_bpermute).
// q: [B,Q,H,DH] f16, kv: [B,K,H,256] f16 (k = e<128, v = e>=128), o: [B,Q,H,DH] f16
// ---------------------------------------------------------------------------
__global__ void __launch_bounds__(256) attn_wmma_kernel(
    const _Float16* __restrict__ qb, const _Float16* __restrict__ kvb,
    _Float16* __restrict__ ob)
{
    __shared__ _Float16 Ks [32][136];       // [kv][dh] (136 = 128 + TDM pad)
    __shared__ _Float16 VsT[128][40];       // [dh][kv]  (transposed)
    __shared__ _Float16 Pst[8][16][40];     // per-wave P staging [m][n]

    const int tid  = threadIdx.x;
    const int wave = tid >> 5, lane = tid & 31;
    const int ln   = lane & 15, hi = lane >> 4;
    const int qblk = blockIdx.x & 3;
    const int h    = (blockIdx.x >> 2) & 7;
    const int b    = blockIdx.x >> 5;
    const int q0   = qblk * 128 + wave * 16;

    // Q fragments (A layout), 4 chunks of dh=32
    v16h qf[4];
#pragma unroll
    for (int f = 0; f < 4; ++f) {
        const _Float16* qrow = qb + (((size_t)(b * Qq + q0 + ln)) * Hh + h) * Dh + f * 32 + hi * 8;
        qf[f] = make_frag(qrow, qrow + 16);
    }

    v8f Oa[8];
#pragma unroll
    for (int t = 0; t < 8; ++t) Oa[t] = (v8f){0,0,0,0,0,0,0,0};
    float mst[8], lst[8];
#pragma unroll
    for (int r = 0; r < 8; ++r) { mst[r] = -1e30f; lst[r] = 0.f; }

    const float scale = 0.08838834764831845f;  // 1/sqrt(128)

    for (int c0 = 0; c0 < Kn; c0 += 32) {
#if __has_builtin(__builtin_amdgcn_tensor_load_to_lds)
        // --- K chunk via Tensor Data Mover (wave 0 issues one 2D descriptor) ---
        if (tid < 32) {
            const unsigned long long gaddr =
                (unsigned long long)(uintptr_t)kvb +
                2ull * ((((unsigned long long)(b * Kn + c0)) * Hh + h) * 256);
            const unsigned int laddr = (unsigned int)(uintptr_t)(&Ks[0][0]);
            u32x4 g0;
            g0[0] = 1u;                                        // count=1 valid descriptor
            g0[1] = laddr;                                     // LDS byte address
            g0[2] = (unsigned int)gaddr;                       // global_addr[31:0]
            g0[3] = (unsigned int)((gaddr >> 32) & 0x01FFFFFFu) | (2u << 30);  // type=2
            i32x8 g1;
            g1[0] = (1 << 16) | (1 << 20) | (5 << 22) | (3 << 25); // data_size=2B, pad_en, 64dw interval, 4dw pad
            g1[1] = (int)(128u << 16);   // tensor_dim0 = 128 (lo16 in [31:16])
            g1[2] = (int)(32u << 16);    // tensor_dim1 = 32  (lo16 in [31:16])
            g1[3] = (int)(128u << 16);   // tile_dim0 = 128
            g1[4] = 32;                  // tile_dim1 = 32, tile_dim2 = 0
            g1[5] = (int)(Hh * 256);     // tensor_dim0_stride = 2048 (lo32)
            g1[6] = 0;                   // stride hi / dim1_stride lo
            g1[7] = 0;
            i32x4 g2; g2[0] = 0; g2[1] = 0; g2[2] = 0; g2[3] = 0;
            i32x4 g3; g3[0] = 0; g3[1] = 0; g3[2] = 0; g3[3] = 0;
#if __clang_major__ >= 23
            i32x8 g4;
#pragma unroll
            for (int z = 0; z < 8; ++z) g4[z] = 0;
            __builtin_amdgcn_tensor_load_to_lds(g0, g1, g2, g3, g4, 0);
#else
            __builtin_amdgcn_tensor_load_to_lds(g0, g1, g2, g3, 0);
#endif
            wait_tensorcnt0();
        }
        // --- V chunk: cooperative transposed copy ---
#pragma unroll
        for (int l = 0; l < 2; ++l) {
            int e = (l * 256 + tid) * 8;
            int r = e >> 7, cc = e & 127;
            const _Float16* src = kvb + (((size_t)(b * Kn + c0 + r)) * Hh + h) * 256;
            v8h vv = *reinterpret_cast<const v8h*>(src + 128 + cc);
#pragma unroll
            for (int u = 0; u < 8; ++u) VsT[cc + u][r] = vv[u];
        }
#else
        // Fallback: cooperative K+V copy through VGPRs
#pragma unroll
        for (int l = 0; l < 2; ++l) {
            int e = (l * 256 + tid) * 8;
            int r = e >> 7, cc = e & 127;
            const _Float16* src = kvb + (((size_t)(b * Kn + c0 + r)) * Hh + h) * 256;
            v8h kk = *reinterpret_cast<const v8h*>(src + cc);
            *reinterpret_cast<v8h*>(&Ks[r][cc]) = kk;
            v8h vv = *reinterpret_cast<const v8h*>(src + 128 + cc);
#pragma unroll
            for (int u = 0; u < 8; ++u) VsT[cc + u][r] = vv[u];
        }
#endif
        __syncthreads();

        // S = Q * K^T  (two 16x16 tiles: kv 0..15 and 16..31)
        v8f sL = {0,0,0,0,0,0,0,0};
        v8f sR = {0,0,0,0,0,0,0,0};
#pragma unroll
        for (int f = 0; f < 4; ++f) {
            const _Float16* kL = &Ks[ln][f * 32];
            const _Float16* kR = &Ks[16 + ln][f * 32];
            v16h bL = make_frag(kL + hi * 8, kL + 16 + hi * 8);
            v16h bR = make_frag(kR + hi * 8, kR + 16 + hi * 8);
            sL = __builtin_amdgcn_wmma_f32_16x16x32_f16(false, qf[f], false, bL, (short)0, sL, false, false);
            sR = __builtin_amdgcn_wmma_f32_16x16x32_f16(false, qf[f], false, bR, (short)0, sR, false, false);
        }

        // Online softmax; VGPR r holds row (r + 8*hi), its 16 cols live in a
        // 16-lane half -> DPP16 butterfly reductions, no LDS traffic.
#pragma unroll
        for (int r = 0; r < 8; ++r) {
            float a  = sL[r] * scale;
            float b2 = sR[r] * scale;
            float v = red_max16(fmaxf(a, b2));
            float mnew = fmaxf(mst[r], v);
            float corr = __expf(mst[r] - mnew);
            float pL = __expf(a - mnew);
            float pR = __expf(b2 - mnew);
            float rs = red_add16(pL + pR);
            lst[r] = lst[r] * corr + rs;
            mst[r] = mnew;
#pragma unroll
            for (int t = 0; t < 8; ++t) Oa[t][r] *= corr;
            Pst[wave][r + hi * 8][ln]      = (_Float16)pL;
            Pst[wave][r + hi * 8][16 + ln] = (_Float16)pR;
        }

        // O += P * V  (A = P 16x32, B = V 32x16 per dh col-tile from VsT)
        const _Float16* prow = &Pst[wave][ln][0];
        v16h pa = make_frag(prow + hi * 8, prow + 16 + hi * 8);
#pragma unroll
        for (int t = 0; t < 8; ++t) {
            const _Float16* vrow = &VsT[t * 16 + ln][0];
            v16h bv = make_frag(vrow + hi * 8, vrow + 16 + hi * 8);
            Oa[t] = __builtin_amdgcn_wmma_f32_16x16x32_f16(false, pa, false, bv, (short)0, Oa[t], false, false);
        }
        __syncthreads();
    }

    // Normalize and write o (f16)
#pragma unroll
    for (int r = 0; r < 8; ++r) {
        float inv = 1.0f / lst[r];
        size_t qr = (size_t)(q0 + r + hi * 8);
#pragma unroll
        for (int t = 0; t < 8; ++t) {
            ob[((b * Qq + qr) * Hh + h) * Dh + t * 16 + ln] = (_Float16)(Oa[t][r] * inv);
        }
    }
}

// ---------------------------------------------------------------------------
extern "C" void kernel_launch(void* const* d_in, const int* in_sizes, int n_in,
                              void* d_out, int out_size, void* d_ws, size_t ws_size,
                              hipStream_t stream) {
    (void)in_sizes; (void)n_in; (void)out_size; (void)ws_size;

    const float* inputs_q  = (const float*)d_in[0];
    const float* inputs_kv = (const float*)d_in[1];
    const float* Wq        = (const float*)d_in[2];
    const float* Wkv       = (const float*)d_in[3];
    const float* Wproj     = (const float*)d_in[4];
    const float* b_proj    = (const float*)d_in[5];

    char* ws = (char*)d_ws;
    _Float16* qh  = (_Float16*)ws;                                              // 2048*1024 f16 = 4 MB
    _Float16* kvh = (_Float16*)(ws + (size_t)2048 * 1024 * 2);                  // 16384*2048 f16 = 64 MB
    _Float16* oh  = (_Float16*)(ws + (size_t)2048 * 1024 * 2
                                    + (size_t)16384 * 2048 * 2);                // 2048*1024 f16 = 4 MB

    // Q projection: [2048 x 1024] x [1024 x 1024] -> f16
    gemm_wmma_kernel<float, true, false><<<dim3(1024 / 128, 2048 / 128), 256, 0, stream>>>(
        inputs_q, Wq, (void*)qh, nullptr, Bb * Qq, Hh * Dh, Dm);
    // KV projection: [16384 x 1024] x [1024 x 2048] -> f16
    gemm_wmma_kernel<float, true, false><<<dim3(2048 / 128, 16384 / 128), 256, 0, stream>>>(
        inputs_kv, Wkv, (void*)kvh, nullptr, Bb * Kn, Hh * 2 * Dh, Dm);
    // Fused flash attention
    attn_wmma_kernel<<<dim3(Bb * Hh * (Qq / 128)), 256, 0, stream>>>(qh, kvh, oh);
    // Output projection + bias: [2048 x 1024] x [1024 x 1024] -> f32 out
    gemm_wmma_kernel<_Float16, false, true><<<dim3(1024 / 128, 2048 / 128), 256, 0, stream>>>(
        oh, Wproj, d_out, b_proj, Bb * Qq, Dm, Hh * Dh);
}
